// RecModel_72086731096654
// MI455X (gfx1250) — matrix operand order repaired and checked by reading, not compiled, not run on hardware
//
#include <hip/hip_runtime.h>
#include <hip/hip_bf16.h>
#include <math.h>

typedef __attribute__((ext_vector_type(16))) __bf16 v16bf;
typedef __attribute__((ext_vector_type(8)))  __bf16 v8bf;
typedef __attribute__((ext_vector_type(8)))  float  v8f;
typedef __attribute__((ext_vector_type(4)))  unsigned int v4u;
typedef __attribute__((ext_vector_type(8)))  int    v8i;
typedef __attribute__((ext_vector_type(4)))  int    v4i;

typedef __attribute__((address_space(3))) float lds_f32;

#define N_ENTITY 30000
#define D        128
#define R        12
#define NB       4
#define E_EDGES  200000
#define BATCH    256
#define L        64
#define NR       (N_ENTITY * R)        // 360000 histogram bins

// workspace layout (byte offsets)
#define WS_COUNTS 0u                   // 360000 f32 = 1.44 MB
#define WS_KG     0x180000u            // 30000*128 f32 = 15.36 MB
#define WS_KGB    0x1100000u           // 30000*128 bf16 = 7.68 MB
#define WS_USERB  0x1900000u           // 256*128 bf16 = 64 KB

// ---------------- RGCN stage ----------------

__global__ void k_zero(float* counts, float* loss_slot) {
    int i = blockIdx.x * 256 + threadIdx.x;
    if (i < NR) counts[i] = 0.0f;
    if (i == 0) *loss_slot = 0.0f;
}

__global__ void k_hist(const int* ei, const int* et, float* counts) {
    int e = blockIdx.x * 256 + threadIdx.x;
    if (e >= E_EDGES) return;
    int dst = ei[E_EDGES + e];
    int t   = et[e];
    atomicAdd(&counts[dst * R + t], 1.0f);
}

__global__ void k_kg_init(const float* root, const float* bias, float* kg) {
    int i = blockIdx.x * 256 + threadIdx.x;
    if (i < N_ENTITY * D) kg[i] = root[i] + bias[i & (D - 1)];
}

__global__ void k_edges(const int* ei, const int* et, const float* bases,
                        const float* comp, const float* counts, float* kg) {
    int idx = blockIdx.x * 256 + threadIdx.x;
    int e = idx >> 7;           // edge
    int d = idx & 127;          // feature
    if (e >= E_EDGES) return;
    int src = ei[e];
    int dst = ei[E_EDGES + e];
    int t   = et[e];
    float c0 = comp[t * NB + 0];
    float c1 = comp[t * NB + 1];
    float c2 = comp[t * NB + 2];
    float c3 = comp[t * NB + 3];
    float norm = 1.0f / fmaxf(counts[dst * R + t], 1.0f);
    const size_t plane = (size_t)N_ENTITY * D;
    const float* bp = bases + (size_t)src * D + d;
    float v = c0 * bp[0] + c1 * bp[plane] + c2 * bp[2 * plane] + c3 * bp[3 * plane];
    atomicAdd(&kg[(size_t)dst * D + d], v * norm);
}

__global__ void k_tobf(const float* kg, __bf16* kgb) {
    int i = blockIdx.x * 256 + threadIdx.x;
    if (i < N_ENTITY * D) kgb[i] = (__bf16)kg[i];
}

// ---------------- Attention stage (WMMA #1 + TDM) ----------------
// One block per batch row b. h(64x128) @ Wa(128x128) -> tanh -> dot va -> softmax -> user.
// Wa is DMA'd into LDS by the Tensor Data Mover while the block gathers h.

#if __has_builtin(__builtin_amdgcn_tensor_load_to_lds) && \
    __has_builtin(__builtin_amdgcn_s_wait_tensorcnt)
#define HAVE_TDM 1
#if __has_include(<hip/amd_detail/amd_gfx1250_TDM.h>)
// therock/clang-23 toolchain: 6-arg form (extra int32x8 group)
#define TDM_LOAD(g0, g1, z4a, z4b, z8) \
    __builtin_amdgcn_tensor_load_to_lds(g0, g1, z4a, z4b, z8, 0)
#else
// ROCm 7.2 / clang-22: 5-arg form
#define TDM_LOAD(g0, g1, z4a, z4b, z8) \
    __builtin_amdgcn_tensor_load_to_lds(g0, g1, z4a, z4b, 0)
#endif
#else
#define HAVE_TDM 0
#endif

__global__ __launch_bounds__(256) void k_attn(const int* entities, const float* sent,
                                              const float* kg, const float* Wa,
                                              const float* ba, const float* va,
                                              __bf16* userb) {
    __shared__ __align__(128) float  s_wa[D * D];   // 64 KB, TDM destination
    __shared__ __align__(32)  float  s_h[L * D];    // 32 KB
    __shared__ __align__(32)  __bf16 s_hb[L * D];   // 16 KB
    __shared__ __align__(32)  __bf16 s_waT[D * D];  // 32 KB (Wa transposed, bf16)
    __shared__ float s_scores[L];
    __shared__ float s_alpha[L];

    int b = blockIdx.x;
    int t = threadIdx.x;
    int wv = t >> 5, ln = t & 31;

#if HAVE_TDM
    // Wave 0 issues a TDM 2-D tile load: Wa (128x128 f32, contiguous) -> s_wa.
    if (wv == 0) {
        unsigned long long ga = (unsigned long long)(const void*)Wa;
        unsigned lds_off = (unsigned)(unsigned long long)(lds_f32*)s_wa;
        // D# group0: count=1 | lds_addr | global_addr[56:0] | type=2
        v4u g0 = { 1u,
                   lds_off,
                   (unsigned)(ga & 0xFFFFFFFFu),
                   (unsigned)(((ga >> 32) & 0x01FFFFFFu) | (2u << 30)) };
        // D# group1: data_size=2(4B); tensor_dim0=128; tensor_dim1=128;
        //            tile_dim0=128; tile_dim1=128; tensor_dim0_stride=128
        v8i g1 = { (int)(2u << 16),       // wg_mask=0, data_size=4B
                   (int)(128u << 16),     // tensor_dim0[15:0] in bits 63:48
                   (int)(128u << 16),     // tensor_dim1[15:0] in bits 95:80
                   (int)(128u << 16),     // tile_dim0 in bits 127:112
                   128,                   // tile_dim1 in bits 143:128
                   128,                   // tensor_dim0_stride[31:0]
                   0, 0 };
        v4i z4 = { 0, 0, 0, 0 };
        v8i z8 = { 0, 0, 0, 0, 0, 0, 0, 0 };
        TDM_LOAD(g0, g1, z4, z4, z8);
    }
#else
    for (int i = t; i < D * D; i += 256) s_wa[i] = Wa[i];
#endif

    // Overlap with the DMA: gather h = kg[entities] * sent (f32 + bf16 copies).
    for (int i = t; i < L * D; i += 256) {
        int l = i >> 7, d = i & 127;
        int ent = entities[b * L + l];
        float hv = kg[(size_t)ent * D + d] * sent[b * L + l];
        s_h[i]  = hv;
        s_hb[i] = (__bf16)hv;
    }
    if (t < L) s_scores[t] = 0.0f;

#if HAVE_TDM
    if (wv == 0) __builtin_amdgcn_s_wait_tensorcnt(0);  // issuing wave drains TENSORcnt
#endif
    __syncthreads();

    // transpose + convert: s_waT[n*D + k] = (bf16) Wa[k*D + n]
    for (int i = t; i < D * D; i += 256) {
        int k = i >> 7, n = i & 127;
        s_waT[n * D + k] = (__bf16)s_wa[i];
    }
    __syncthreads();

    // 8 waves, each: M-tile mt = wv>>1, N-tiles [ (wv&1)*4 , +4 )
    int mt = wv >> 1;
    int ngrp = (wv & 1) * 4;
    int hi = ln >> 4, lo = ln & 15;
    int Mrow = mt * 16 + lo;

    // preload A fragments for 4 K-steps (K=32 each)
    v16bf afr[4];
#pragma unroll
    for (int ks = 0; ks < 4; ks++) {
        const __bf16* ap = &s_hb[Mrow * D + ks * 32 + hi * 16];
        v8bf a0 = *(const v8bf*)ap;
        v8bf a1 = *(const v8bf*)(ap + 8);
#pragma unroll
        for (int i = 0; i < 8; i++) { afr[ks][i] = a0[i]; afr[ks][i + 8] = a1[i]; }
    }

    for (int nt = ngrp; nt < ngrp + 4; nt++) {
        int Ncol = nt * 16 + lo;
        v8f acc = {};
#pragma unroll
        for (int ks = 0; ks < 4; ks++) {
            v16bf bfr = *(const v16bf*)&s_waT[Ncol * D + ks * 32 + hi * 16];
            acc = __builtin_amdgcn_wmma_f32_16x16x32_bf16(
                false, afr[ks], false, bfr, (short)0, acc, false, false);
        }
        float ban = ba[Ncol], van = va[Ncol];
#pragma unroll
        for (int r = 0; r < 8; r++) {
            int row = mt * 16 + r + hi * 8;
            atomicAdd(&s_scores[row], tanhf(acc[r] + ban) * van);
        }
    }
    __syncthreads();

    // masked softmax over L (scalar on thread 0 — L is tiny)
    if (t == 0) {
        float mx = -1e30f; int any = 0;
        for (int l = 0; l < L; l++) {
            if (entities[b * L + l] != 0) { any = 1; mx = fmaxf(mx, s_scores[l]); }
        }
        float sum = 0.0f;
        for (int l = 0; l < L; l++) {
            float a = 0.0f;
            if (entities[b * L + l] != 0) a = __expf(s_scores[l] - mx);
            s_alpha[l] = a; sum += a;
        }
        float inv = any ? (1.0f / sum) : 0.0f;
        for (int l = 0; l < L; l++) s_alpha[l] *= inv;
    }
    __syncthreads();

    // user[b,d] = sum_l alpha[l] * h[l,d]
    if (t < D) {
        float acc = 0.0f;
        for (int l = 0; l < L; l++) acc += s_alpha[l] * s_h[l * D + t];
        userb[b * D + t] = (__bf16)acc;
    }
}

// ---------------- Rec-score GEMM (WMMA #2) ----------------
// out[m,n] = sum_k user[m,k]*kg[n,k] + rec_bias[n].
// Block: 8 waves; wave w computes a 16x64 strip (4 N-tiles), reusing A fragments.

__global__ __launch_bounds__(256) void k_rec(const __bf16* userb, const __bf16* kgb,
                                             const float* rec_bias, float* out) {
    int t = threadIdx.x;
    int wv = t >> 5, ln = t & 31;
    int hi = ln >> 4, lo = ln & 15;
    int m0 = blockIdx.y * 16;
    int nbase = blockIdx.x * 512 + wv * 64;

    // A fragments: user rows m0..m0+15, all 4 K-steps (reused across 4 N-tiles)
    v16bf afr[4];
    const __bf16* arow = userb + (m0 + lo) * D;
#pragma unroll
    for (int ks = 0; ks < 4; ks++) {
        const __bf16* ap = arow + ks * 32 + hi * 16;
        v8bf a0 = *(const v8bf*)ap;
        v8bf a1 = *(const v8bf*)(ap + 8);
#pragma unroll
        for (int i = 0; i < 8; i++) { afr[ks][i] = a0[i]; afr[ks][i + 8] = a1[i]; }
    }

#pragma unroll
    for (int nt = 0; nt < 4; nt++) {
        int ncol = nbase + nt * 16 + lo;
        int nrow = ncol < N_ENTITY ? ncol : (N_ENTITY - 1);   // clamp loads, guard stores
        const __bf16* brow = kgb + (size_t)nrow * D;          // contraction axis contiguous

        v8f acc = {};
#pragma unroll
        for (int ks = 0; ks < 4; ks++) {
            v16bf bfr = *(const v16bf*)(brow + ks * 32 + hi * 16);
            acc = __builtin_amdgcn_wmma_f32_16x16x32_bf16(
                false, afr[ks], false, bfr, (short)0, acc, false, false);
        }

        if (ncol < N_ENTITY) {
            float bias = rec_bias[ncol];
#pragma unroll
            for (int r = 0; r < 8; r++) {
                out[(size_t)(m0 + r + hi * 8) * N_ENTITY + ncol] = acc[r] + bias;
            }
        }
    }
}

// ---------------- log-softmax loss ----------------

__global__ __launch_bounds__(256) void k_loss(const float* scores, const int* labels,
                                              float* loss) {
    __shared__ float red[256];
    int b = blockIdx.x, t = threadIdx.x;
    const float* row = scores + (size_t)b * N_ENTITY;

    float mx = -1e30f;
    for (int i = t; i < N_ENTITY; i += 256) mx = fmaxf(mx, row[i]);
    red[t] = mx; __syncthreads();
    for (int s = 128; s > 0; s >>= 1) {
        if (t < s) red[t] = fmaxf(red[t], red[t + s]);
        __syncthreads();
    }
    mx = red[0]; __syncthreads();

    float sum = 0.0f;
    for (int i = t; i < N_ENTITY; i += 256) sum += __expf(row[i] - mx);
    red[t] = sum; __syncthreads();
    for (int s = 128; s > 0; s >>= 1) {
        if (t < s) red[t] += red[t + s];
        __syncthreads();
    }
    if (t == 0) {
        float lse = mx + __logf(red[0]);
        atomicAdd(loss, (lse - row[labels[b]]) * (1.0f / BATCH));
    }
}

// ---------------- launch ----------------

extern "C" void kernel_launch(void* const* d_in, const int* in_sizes, int n_in,
                              void* d_out, int out_size, void* d_ws, size_t ws_size,
                              hipStream_t stream) {
    const int*   entities   = (const int*)d_in[0];
    const int*   labels     = (const int*)d_in[1];
    const float* sent       = (const float*)d_in[2];
    const int*   edge_index = (const int*)d_in[3];
    const int*   edge_type  = (const int*)d_in[4];
    const float* bases      = (const float*)d_in[5];
    const float* comp       = (const float*)d_in[6];
    const float* root       = (const float*)d_in[7];
    const float* rgcn_bias  = (const float*)d_in[8];
    const float* Wa         = (const float*)d_in[9];
    const float* ba         = (const float*)d_in[10];
    const float* va         = (const float*)d_in[11];
    const float* rec_bias   = (const float*)d_in[12];

    float* out_scores = (float*)d_out;
    float* out_loss   = out_scores + (size_t)BATCH * N_ENTITY;

    char*   ws     = (char*)d_ws;
    float*  counts = (float*)(ws + WS_COUNTS);
    float*  kg     = (float*)(ws + WS_KG);
    __bf16* kgb    = (__bf16*)(ws + WS_KGB);
    __bf16* userb  = (__bf16*)(ws + WS_USERB);

    k_zero   <<<(NR + 255) / 256, 256, 0, stream>>>(counts, out_loss);
    k_hist   <<<(E_EDGES + 255) / 256, 256, 0, stream>>>(edge_index, edge_type, counts);
    k_kg_init<<<(N_ENTITY * D + 255) / 256, 256, 0, stream>>>(root, rgcn_bias, kg);
    k_edges  <<<(E_EDGES * D) / 256, 256, 0, stream>>>(edge_index, edge_type, bases,
                                                       comp, counts, kg);
    k_tobf   <<<(N_ENTITY * D + 255) / 256, 256, 0, stream>>>(kg, kgb);
    k_attn   <<<BATCH, 256, 0, stream>>>(entities, sent, kg, Wa, ba, va, userb);
    dim3 grec((N_ENTITY + 511) / 512, BATCH / 16);
    k_rec    <<<grec, 256, 0, stream>>>(userb, kgb, rec_bias, out_scores);
    k_loss   <<<BATCH, 256, 0, stream>>>(out_scores, labels, out_loss);
}
